// GPLDA_73229192397052
// MI455X (gfx1250) — matrix operand order repaired and checked by reading, not compiled, not run on hardware
//
#include <hip/hip_runtime.h>

typedef __attribute__((ext_vector_type(16))) _Float16 v16h;
typedef __attribute__((ext_vector_type(8)))  _Float16 v8h;
typedef __attribute__((ext_vector_type(8)))  float    v8f;

#define DD 128
#define NE 8192
#define NT 8192

// ---------------------------------------------------------------------------
// Kernel 1: M = W @ U_k   (128x128 @ 128x128, trivial)
// block i (row), thread k (col): W[i][j] is block-uniform, U[j][k] coalesced.
// ---------------------------------------------------------------------------
__global__ void make_M(const float* __restrict__ W, const float* __restrict__ U,
                       float* __restrict__ M) {
  int i = blockIdx.x;
  int k = threadIdx.x;
  float acc = 0.f;
#pragma unroll 8
  for (int j = 0; j < DD; ++j)
    acc = fmaf(W[i * DD + j], U[j * DD + k], acc);
  M[i * DD + k] = acc;
}

// ---------------------------------------------------------------------------
// Kernel 2: per-sample projection + quadratic score.
//   y = (x - miu)/(||x-miu|| + eps);  z = M^T y;  s = z^T Q z
//   zh[n][k] = f16( z[k] * (useLam ? 2*lam[k] : 1) )
// One 128-thread block per sample. All matrix accesses are column reads
// (coalesced across threads). z^T Q^T z == z^T Q z, so column access is exact.
// ---------------------------------------------------------------------------
__global__ __launch_bounds__(DD) void project(
    const float* __restrict__ x, const float* __restrict__ miu,
    const float* __restrict__ M, const float* __restrict__ Q,
    const float* __restrict__ lam,
    _Float16* __restrict__ zh, float* __restrict__ s, int useLam) {
  __shared__ float red[DD];
  __shared__ float ysh[DD];
  __shared__ float zsh[DD];
  const int n = blockIdx.x;
  const int t = threadIdx.x;

  float v = x[(size_t)n * DD + t] - miu[t];
  red[t] = v * v;
  __syncthreads();
#pragma unroll
  for (int off = 64; off > 0; off >>= 1) {
    if (t < off) red[t] += red[t + off];
    __syncthreads();
  }
  const float norm = sqrtf(red[0]) + 1e-8f;
  ysh[t] = v / norm;
  __syncthreads();

  float z = 0.f;
#pragma unroll 8
  for (int d = 0; d < DD; ++d)                 // z[t] = sum_d M[d][t]*y[d]
    z = fmaf(M[d * DD + t], ysh[d], z);
  zsh[t] = z;
  __syncthreads();

  float q = 0.f;
#pragma unroll 8
  for (int j = 0; j < DD; ++j)                 // q[t] = (Q^T z)[t]
    q = fmaf(Q[j * DD + t], zsh[j], q);

  red[t] = z * q;
  __syncthreads();
#pragma unroll
  for (int off = 64; off > 0; off >>= 1) {
    if (t < off) red[t] += red[t + off];
    __syncthreads();
  }
  if (t == 0) s[n] = red[0];

  const float scale = useLam ? 2.0f * lam[t] : 1.0f;
  zh[(size_t)n * DD + t] = (_Float16)(z * scale);
}

// ---------------------------------------------------------------------------
// Kernel 3: out[n,m] = sum_k A[n,k]*B[m,k] + s1[n] + s2[m]
// A = enroll operand (2*lam folded), B = test operand, both [sample][128] f16.
// One 16x16 tile per wave, 8 waves per block; K=128 -> 4 WMMA 16x16x32 f16.
//
// Per-lane operand layouts (CDNA5 ISA 7.12.2, wave32):
//  A 16x32:  lane L (row M=L%16, hi=L/16):
//            VGPR0..3 = K[k0 + hi*8 .. +7], VGPR4..7 = K[k0+16+hi*8 .. +7]
//            -> two 16-byte loads from the row-major A row.
//  B 32x16:  lane L (col N=L%16, hi=L/16):
//            VGPR0..3 = K[k0 + hi*8 .. +7], VGPR4..7 = K[k0+16+hi*8 .. +7]
//            -> two 16-byte loads from the row-major B row.
//  C/D 16x16 f32: VGPR r, lane L -> (row = r + 8*hi, col = L%16).
// ---------------------------------------------------------------------------
__global__ __launch_bounds__(256) void score_gemm(
    const _Float16* __restrict__ A, const _Float16* __restrict__ B,
    const float* __restrict__ s1, const float* __restrict__ s2,
    float* __restrict__ out) {
  const int wave = threadIdx.x >> 5;
  const int lane = threadIdx.x & 31;
  const int n0 = blockIdx.y * 16;                  // M tile
  const int m0 = (blockIdx.x * 8 + wave) * 16;     // N tile
  const int r  = lane & 15;
  const int hi = lane >> 4;

  const _Float16* ap = A + (size_t)(n0 + r) * DD + hi * 8;
  const _Float16* bp = B + (size_t)(m0 + r) * DD + hi * 8;

  v8f acc = {};
#pragma unroll
  for (int kk = 0; kk < 4; ++kk) {                 // k0 = kk*32
    union { v16h v; v8h h[2]; } av;
    av.h[0] = *(const v8h*)(ap + kk * 32);         // K = k0 + hi*8 .. +7
    av.h[1] = *(const v8h*)(ap + kk * 32 + 16);    // K = k0 + 16 + hi*8 .. +7
    union { v16h v; v8h h[2]; } bu;
    bu.h[0] = *(const v8h*)(bp + kk * 32);         // K = k0 + hi*8 .. +7
    bu.h[1] = *(const v8h*)(bp + kk * 32 + 16);    // K = k0 + 16 + hi*8 .. +7
    const v16h bv = bu.v;
    acc = __builtin_amdgcn_wmma_f32_16x16x32_f16(
        false, av.v, false, bv, (short)0, acc, false, false);
  }

  const float s2v = s2[m0 + r];
#pragma unroll
  for (int rr = 0; rr < 8; ++rr) {
    const int row = n0 + rr + hi * 8;
    const float val = acc[rr] + s1[row] + s2v;
    __builtin_nontemporal_store(val, &out[(size_t)row * NT + (m0 + r)]);
  }
}

// ---------------------------------------------------------------------------
extern "C" void kernel_launch(void* const* d_in, const int* in_sizes, int n_in,
                              void* d_out, int out_size, void* d_ws, size_t ws_size,
                              hipStream_t stream) {
  const float* x_enroll = (const float*)d_in[0];
  const float* x_test   = (const float*)d_in[1];
  const float* W        = (const float*)d_in[2];
  const float* miu      = (const float*)d_in[3];
  const float* lam      = (const float*)d_in[4];
  const float* U_k      = (const float*)d_in[5];
  const float* Q_hat    = (const float*)d_in[6];
  float* out            = (float*)d_out;

  char* ws = (char*)d_ws;
  float*    M  = (float*)ws;                                       //  64 KB
  _Float16* Ae = (_Float16*)(ws + (64u << 10));                    //   2 MB
  _Float16* At = (_Float16*)(ws + (64u << 10) + (2u << 20));       //   2 MB
  float*    s1 = (float*)(ws + (64u << 10) + (4u << 20));          //  32 KB
  float*    s2 = (float*)(ws + (64u << 10) + (4u << 20) + (32u << 10));

  make_M<<<dim3(DD), dim3(DD), 0, stream>>>(W, U_k, M);
  project<<<dim3(NE), dim3(DD), 0, stream>>>(x_enroll, miu, M, Q_hat, lam, Ae, s1, 1);
  project<<<dim3(NT), dim3(DD), 0, stream>>>(x_test,   miu, M, Q_hat, lam, At, s2, 0);
  score_gemm<<<dim3(NT / 128, NE / 16), 256, 0, stream>>>(Ae, At, s1, s2, out);
}
